// Net_59021440582334
// MI455X (gfx1250) — matrix operand order repaired and checked by reading, not compiled, not run on hardware
//
#include <hip/hip_runtime.h>

// ---------------------------------------------------------------------------
// GCN (3x GCNConv + residual) + MLP head for MI455X (gfx1250, wave32, WMMA)
// ---------------------------------------------------------------------------
//  * GEMMs (5x 50000x128x128 + 1x 50000x128x48pad) via v_wmma_f32_16x16x32_bf16
//    (bf16 inputs, fp32 accumulate). N-tile count is a template parameter so
//    the WMMA loop is branch-free; B fragments software-pipelined by 2 so
//    LDS loads overlap WMMA issue.
//  * Aggregation via CSR gather (one wave32 per node). The 25.6MB feature
//    matrix is L2-resident (192MB L2); gathers hit L2, no float atomics.
//    global_prefetch_b8 on the next neighbor row hides L2 latency.
// ---------------------------------------------------------------------------

typedef __attribute__((ext_vector_type(16))) __bf16 v16bf;
typedef __attribute__((ext_vector_type(8)))  float  v8f;

union FragAB { v16bf v; unsigned u[8]; };

#define NN 50000
#define NE 800000
#define HID 128
#define NCLS 40
#define NCLSP 48
#define NL 3

// ---------------- small utility kernels ----------------

__global__ void cvt_f32_bf16_kernel(const float* __restrict__ in,
                                    __bf16* __restrict__ out, int n) {
  int i = blockIdx.x * 256 + threadIdx.x;
  if (i < n) out[i] = (__bf16)in[i];
}

__global__ void pad_fc3w_kernel(const float* __restrict__ w /*128x40*/,
                                __bf16* __restrict__ out /*128x48*/) {
  int idx = blockIdx.x * 256 + threadIdx.x;
  if (idx >= HID * NCLSP) return;
  int k = idx / NCLSP, n = idx % NCLSP;
  out[idx] = (n < NCLS) ? (__bf16)w[k * NCLS + n] : (__bf16)0.0f;
}

__global__ void pad_bias_kernel(const float* __restrict__ b /*40*/,
                                float* __restrict__ out /*48*/) {
  int i = threadIdx.x;
  if (i < NCLSP) out[i] = (i < NCLS) ? b[i] : 0.0f;
}

__global__ void zero_i32_kernel(int* __restrict__ p, int n) {
  int i = blockIdx.x * 256 + threadIdx.x;
  if (i < n) p[i] = 0;
}

// ---------------- CSR construction ----------------

__global__ void count_edges_kernel(const int* __restrict__ ei, int* __restrict__ cnt) {
  int e = blockIdx.x * 256 + threadIdx.x;
  if (e < NE) atomicAdd(&cnt[ei[NE + e]], 1);
}

// single-workgroup Hillis-Steele chunked scan over 50000 counts
__global__ void scan_kernel(const int* __restrict__ cnt, int* __restrict__ rowptr,
                            int* __restrict__ cursor) {
  __shared__ int s[1024];
  __shared__ int sbase;
  int t = threadIdx.x;
  if (t == 0) sbase = 0;
  __syncthreads();
  for (int base = 0; base < NN; base += 1024) {
    int i = base + t;
    int v = (i < NN) ? cnt[i] : 0;
    s[t] = v;
    __syncthreads();
#pragma unroll
    for (int off = 1; off < 1024; off <<= 1) {
      int add = (t >= off) ? s[t - off] : 0;
      __syncthreads();
      s[t] += add;
      __syncthreads();
    }
    int excl = s[t] - v;
    if (i < NN) { rowptr[i] = sbase + excl; cursor[i] = sbase + excl; }
    __syncthreads();
    if (t == 0) sbase += s[1023];
    __syncthreads();
  }
  if (t == 0) rowptr[NN] = sbase;
}

__global__ void fill_csr_kernel(const int* __restrict__ ei, int* __restrict__ cursor,
                                int* __restrict__ colb) {
  int e = blockIdx.x * 256 + threadIdx.x;
  if (e >= NE) return;
  int s = ei[e], d = ei[NE + e];
  int pos = atomicAdd(&cursor[d], 1);
  colb[pos] = s;
}

__global__ void norm_kernel(const int* __restrict__ cnt, float* __restrict__ nrm) {
  int i = blockIdx.x * 256 + threadIdx.x;
  if (i < NN) nrm[i] = rsqrtf(1.0f + (float)cnt[i]);  // deg = indeg + self loop
}

// ---------------- WMMA GEMM: H[nrows x NT*16] = Y[nrows x 128] @ W[128 x NT*16] ----
// Block: 256 threads = 8 waves; each block does 128 rows, each wave a 16-row strip.
// NT (number of 16-wide N tiles) is compile-time: branch-free WMMA inner loops.

template <int NT>
__global__ void gemm_wmma_kernel(const __bf16* __restrict__ Y,
                                 const __bf16* __restrict__ W, int nrows, int ldw,
                                 const float* __restrict__ bias, int relu,
                                 float* __restrict__ outF, int ldF, int ncolsF,
                                 __bf16* __restrict__ outB) {
  __shared__ __bf16 sY[128 * 128];  // 32KB
  __shared__ __bf16 sW[128 * 128];  // 32KB ([n][k] transposed)
  int t = threadIdx.x;
  int row0 = blockIdx.x * 128;

  for (int idx = t; idx < NT * 16 * 128; idx += 256) {
    int n = idx >> 7, k = idx & 127;
    sW[idx] = W[k * ldw + n];
  }
  for (int idx = t; idx < 128 * 128; idx += 256) {
    int r = idx >> 7, k = idx & 127;
    int gr = row0 + r;
    sY[idx] = (gr < nrows) ? Y[(size_t)gr * 128 + k] : (__bf16)0.0f;
  }
  __syncthreads();

  int lane = t & 31, wave = t >> 5;
  int half = lane >> 4, mr = lane & 15;

  v8f acc[NT];
#pragma unroll
  for (int i = 0; i < NT; i++) acc[i] = (v8f){0.f, 0.f, 0.f, 0.f, 0.f, 0.f, 0.f, 0.f};

  const unsigned* sYu = (const unsigned*)sY;
  const unsigned* sWu = (const unsigned*)sW;
  int arow = wave * 16 + mr;

#pragma unroll
  for (int kt = 0; kt < 4; kt++) {
    int kb = kt * 32;
    // A fragment: 16x32 bf16 (contiguous k order: k = 16*half + s, matching B)
    FragAB a;
#pragma unroll
    for (int v = 0; v < 8; v++) {
      int k = kb + 16 * half + 2 * v;
      a.u[v] = sYu[(arow * 128 + k) >> 1];
    }
    // N tiles processed in pairs: both B fragments issue before first WMMA,
    // so the second LDS load overlaps WMMA execution.
#pragma unroll
    for (int nt = 0; nt < NT; nt += 2) {
      const bool has2 = (nt + 1 < NT);
      int n0 = nt * 16 + mr;
      FragAB b0, b1;
#pragma unroll
      for (int v = 0; v < 8; v++) {
        int k = kb + 16 * half + 2 * v;
        b0.u[v] = sWu[(n0 * 128 + k) >> 1];
      }
      if (has2) {
        int n1 = (nt + 1) * 16 + mr;
#pragma unroll
        for (int v = 0; v < 8; v++) {
          int k = kb + 16 * half + 2 * v;
          b1.u[v] = sWu[(n1 * 128 + k) >> 1];
        }
      }
      acc[nt] = __builtin_amdgcn_wmma_f32_16x16x32_bf16(
          false, a.v, false, b0.v, (short)0, acc[nt], false, false);
      if (has2) {
        acc[nt + 1] = __builtin_amdgcn_wmma_f32_16x16x32_bf16(
            false, a.v, false, b1.v, (short)0, acc[nt + 1], false, false);
      }
    }
  }

  // epilogue: C/D layout -> lane<16: M=v, lane>=16: M=8+v, N=lane&15
#pragma unroll
  for (int nt = 0; nt < NT; nt++) {
    int colc = nt * 16 + mr;
    float bval = bias ? bias[colc] : 0.0f;
#pragma unroll
    for (int v = 0; v < 8; v++) {
      int row = row0 + wave * 16 + 8 * half + v;
      float val = acc[nt][v] + bval;
      if (relu) val = fmaxf(val, 0.0f);
      if (row < nrows) {
        if (outF && colc < ncolsF) outF[(size_t)row * ldF + colc] = val;
        if (outB) outB[(size_t)row * 128 + colc] = (__bf16)val;
      }
    }
  }
}

// ---------------- CSR gather aggregation (one wave32 per node) ----------------
// out[n] = relu( sum_{s in N(n)} h[s]*norm[s]*norm[n] + h[n]*norm[n]^2 + bias [+ resid] )

__global__ void aggregate_kernel(const float* __restrict__ hf,
                                 const int* __restrict__ rowptr,
                                 const int* __restrict__ colb,
                                 const float* __restrict__ nrm,
                                 const float* __restrict__ bias,
                                 const float* __restrict__ resid,
                                 float* __restrict__ y1out,
                                 __bf16* __restrict__ ybf) {
  int node = blockIdx.x * 8 + (threadIdx.x >> 5);
  if (node >= NN) return;
  int lane = threadIdx.x & 31;

  float nd = nrm[node];
  float4 hn = ((const float4*)(hf + (size_t)node * 128))[lane];
  float c0 = nd * nd;
  float4 acc;
  acc.x = hn.x * c0; acc.y = hn.y * c0; acc.z = hn.z * c0; acc.w = hn.w * c0;

  int beg = rowptr[node], end = rowptr[node + 1];
  for (int i = beg; i < end; i++) {
    int s = colb[i];
    if (i + 1 < end) {
      // one-edge lookahead: global_prefetch_b8 of next neighbor's feature row
      __builtin_prefetch(hf + (size_t)colb[i + 1] * 128 + lane * 4, 0, 0);
    }
    float c = nrm[s] * nd;
    float4 hv = ((const float4*)(hf + (size_t)s * 128))[lane];
    acc.x += hv.x * c; acc.y += hv.y * c; acc.z += hv.z * c; acc.w += hv.w * c;
  }
  float4 b = ((const float4*)bias)[lane];
  acc.x += b.x; acc.y += b.y; acc.z += b.z; acc.w += b.w;
  if (resid) {
    float4 r = ((const float4*)(resid + (size_t)node * 128))[lane];
    acc.x += r.x; acc.y += r.y; acc.z += r.z; acc.w += r.w;
  }
  acc.x = fmaxf(acc.x, 0.0f); acc.y = fmaxf(acc.y, 0.0f);
  acc.z = fmaxf(acc.z, 0.0f); acc.w = fmaxf(acc.w, 0.0f);

  if (y1out) ((float4*)(y1out + (size_t)node * 128))[lane] = acc;
  __bf16* o = ybf + (size_t)node * 128 + lane * 4;
  o[0] = (__bf16)acc.x; o[1] = (__bf16)acc.y;
  o[2] = (__bf16)acc.z; o[3] = (__bf16)acc.w;
}

// ---------------- log_softmax over 40 classes (one wave32 per row) ----------------

__global__ void log_softmax_kernel(const float* __restrict__ logits, float* __restrict__ out) {
  int row = blockIdx.x * 8 + (threadIdx.x >> 5);
  if (row >= NN) return;
  int lane = threadIdx.x & 31;
  const float* r = logits + (size_t)row * NCLS;
  float v0 = r[lane];  // lane < 32 < 40 always valid
  float v1 = (lane < NCLS - 32) ? r[lane + 32] : -3.402823e38f;
  float m = fmaxf(v0, v1);
#pragma unroll
  for (int off = 16; off > 0; off >>= 1) m = fmaxf(m, __shfl_xor(m, off, 32));
  float s = expf(v0 - m) + ((lane < NCLS - 32) ? expf(v1 - m) : 0.0f);
#pragma unroll
  for (int off = 16; off > 0; off >>= 1) s += __shfl_xor(s, off, 32);
  float l = m + logf(s);
  float* o = out + (size_t)row * NCLS;
  o[lane] = v0 - l;
  if (lane < NCLS - 32) o[lane + 32] = v1 - l;
}

// ---------------- host side ----------------

static inline int cdiv(int a, int b) { return (a + b - 1) / b; }

extern "C" void kernel_launch(void* const* d_in, const int* in_sizes, int n_in,
                              void* d_out, int out_size, void* d_ws, size_t ws_size,
                              hipStream_t stream) {
  const float* x     = (const float*)d_in[0];
  const float* Wconv = (const float*)d_in[1];
  const float* bconv = (const float*)d_in[2];
  const float* fc1w  = (const float*)d_in[3];
  const float* fc1b  = (const float*)d_in[4];
  const float* fc2w  = (const float*)d_in[5];
  const float* fc2b  = (const float*)d_in[6];
  const float* fc3w  = (const float*)d_in[7];
  const float* fc3b  = (const float*)d_in[8];
  const int*   ei    = (const int*)d_in[9];
  float* out = (float*)d_out;

  char* ws = (char*)d_ws;
  size_t off = 0;
  auto take = [&](size_t bytes) -> char* {
    char* p = ws + off;
    off = (off + bytes + 255) & ~(size_t)255;
    return p;
  };

  __bf16* wconv_bf = (__bf16*)take((size_t)NL * HID * HID * 2);
  __bf16* wfc1_bf  = (__bf16*)take((size_t)HID * HID * 2);
  __bf16* wfc2_bf  = (__bf16*)take((size_t)HID * HID * 2);
  __bf16* wfc3_bf  = (__bf16*)take((size_t)HID * NCLSP * 2);
  float*  bfc3p    = (float*)take((size_t)NCLSP * 4);
  int*    cnt      = (int*)take((size_t)NN * 4);
  int*    rowptr   = (int*)take((size_t)(NN + 1) * 4);
  int*    cursor   = (int*)take((size_t)NN * 4);
  int*    colb     = (int*)take((size_t)NE * 4);
  float*  nrm      = (float*)take((size_t)NN * 4);
  __bf16* ybf      = (__bf16*)take((size_t)NN * HID * 2);
  __bf16* zbf      = (__bf16*)take((size_t)NN * HID * 2);
  float*  hf       = (float*)take((size_t)NN * HID * 4);
  float*  y1f      = (float*)take((size_t)NN * HID * 4);
  float*  logits   = (float*)take((size_t)NN * NCLS * 4);
  (void)ws_size; (void)in_sizes; (void)n_in; (void)out_size;

  // weight / feature conversion to bf16
  cvt_f32_bf16_kernel<<<cdiv(NN * HID, 256), 256, 0, stream>>>(x, ybf, NN * HID);
  cvt_f32_bf16_kernel<<<cdiv(NL * HID * HID, 256), 256, 0, stream>>>(Wconv, wconv_bf, NL * HID * HID);
  cvt_f32_bf16_kernel<<<cdiv(HID * HID, 256), 256, 0, stream>>>(fc1w, wfc1_bf, HID * HID);
  cvt_f32_bf16_kernel<<<cdiv(HID * HID, 256), 256, 0, stream>>>(fc2w, wfc2_bf, HID * HID);
  pad_fc3w_kernel<<<cdiv(HID * NCLSP, 256), 256, 0, stream>>>(fc3w, wfc3_bf);
  pad_bias_kernel<<<1, 64, 0, stream>>>(fc3b, bfc3p);

  // degrees + CSR
  zero_i32_kernel<<<cdiv(NN, 256), 256, 0, stream>>>(cnt, NN);
  count_edges_kernel<<<cdiv(NE, 256), 256, 0, stream>>>(ei, cnt);
  scan_kernel<<<1, 1024, 0, stream>>>(cnt, rowptr, cursor);
  fill_csr_kernel<<<cdiv(NE, 256), 256, 0, stream>>>(ei, cursor, colb);
  norm_kernel<<<cdiv(NN, 256), 256, 0, stream>>>(cnt, nrm);

  const int GB = cdiv(NN, 128);   // gemm blocks
  const int AB = cdiv(NN, 8);     // aggregate / softmax blocks

  // 3x GCNConv (+bias, +residual-to-layer1, relu) — bias/relu fused in aggregate
  for (int l = 0; l < NL; l++) {
    gemm_wmma_kernel<8><<<GB, 256, 0, stream>>>(ybf, wconv_bf + (size_t)l * HID * HID,
                                                NN, HID, nullptr, 0,
                                                hf, HID, HID, nullptr);
    aggregate_kernel<<<AB, 256, 0, stream>>>(hf, rowptr, colb, nrm,
                                             bconv + (size_t)l * HID,
                                             (l == 0) ? nullptr : y1f,
                                             (l == 0) ? y1f : nullptr, ybf);
  }

  // MLP head
  gemm_wmma_kernel<8><<<GB, 256, 0, stream>>>(ybf, wfc1_bf, NN, HID, fc1b, 1,
                                              nullptr, 0, 0, zbf);
  gemm_wmma_kernel<8><<<GB, 256, 0, stream>>>(zbf, wfc2_bf, NN, HID, fc2b, 1,
                                              nullptr, 0, 0, ybf);
  gemm_wmma_kernel<3><<<GB, 256, 0, stream>>>(ybf, wfc3_bf, NN, NCLSP, bfc3p, 0,
                                              logits, NCLS, NCLS, nullptr);
  log_softmax_kernel<<<AB, 256, 0, stream>>>(logits, out);
}
